// MoELayer_34007551050241
// MI455X (gfx1250) — compile-verified, matrix operand order checked by reading, not run on hardware
//
#include <hip/hip_runtime.h>
#include <math.h>

// ---- problem constants (B=4,S=2048 -> T=8192; H=2048; I=1024; E=8; top-2) ----
#define T_TOK 8192
#define H_DIM 2048
#define I_DIM 1024
#define E_NUM 8
#define MAXTILES (T_TOK / 16) // 512 token-tiles per expert worst case

typedef __attribute__((ext_vector_type(16))) __bf16 v16bf;
typedef __attribute__((ext_vector_type(8)))  float  v8f;
typedef __attribute__((ext_vector_type(4)))  unsigned int u32x4;

union Frag16 { u32x4 q[2]; v16bf v; }; // 32B: one 16x32 bf16 WMMA operand per lane

__device__ __forceinline__ unsigned short f32_bf16(float f) {
  union { float f; unsigned u; } v; v.f = f;
  unsigned u = v.u;
  if ((u & 0x7F800000u) == 0x7F800000u && (u & 0x007FFFFFu))
    return (unsigned short)((u >> 16) | 0x0040u);          // quiet NaN
  return (unsigned short)((u + 0x7FFFu + ((u >> 16) & 1u)) >> 16); // RNE
}

// ---------------- kernel 1: x -> bf16 ----------------
__global__ __launch_bounds__(256) void k_cvt_x(const float* __restrict__ x,
                                               unsigned short* __restrict__ xb, int n) {
  int i = blockIdx.x * 256 + threadIdx.x;
  if (i < n) xb[i] = f32_bf16(x[i]);
}

// ---------------- kernel 2: weight [E][K][N] -> bf16, WMMA-fragment order ----
// dst layout: [e][nt][kt][lane(32)][elem(16)]  (one 16-N x 32-K tile = 512 bf16,
// lane-major so each lane's operand is 2 contiguous b128 loads).
// (k,n) -> (lane,elem) mapping mirrors ISA 7.12.2 "16-bit A-Matrix 16x32" with lanes = N.
__global__ __launch_bounds__(256) void k_swizzle_w(const float* __restrict__ w,
                                                   unsigned short* __restrict__ ws,
                                                   int K, int N) {
  int id  = blockIdx.x * 256 + threadIdx.x;
  int KT = K >> 5, NT = N >> 4;
  int el  = id & 15;
  int L   = (id >> 4) & 31;
  int rest = id >> 9;
  int kt = rest % KT; rest /= KT;
  int nt = rest % NT; int e = rest / NT;
  int n = nt * 16 + (L & 15);
  int k = kt * 32 + ((el >= 8) ? 16 : 0) + ((L >> 4) << 3) + (((el >> 1) & 3) << 1) + (el & 1);
  ws[id] = f32_bf16(w[((size_t)e * K + k) * N + n]);
}

// ---------------- kernel 3: router (one wave32 per token) ----------------
__global__ __launch_bounds__(256) void k_router(const float* __restrict__ x,
                                                const float* __restrict__ gw,
                                                int* __restrict__ cnt,
                                                int* __restrict__ tokList,
                                                float* __restrict__ wtList) {
  int wave = threadIdx.x >> 5, lane = threadIdx.x & 31;
  int t = blockIdx.x * 8 + wave;
  const float* xr = x + (size_t)t * H_DIM;
  float acc[E_NUM];
#pragma unroll
  for (int e = 0; e < E_NUM; e++) acc[e] = 0.f;
  for (int h = lane; h < H_DIM; h += 32) {
    float xv = xr[h];
#pragma unroll
    for (int e = 0; e < E_NUM; e++) acc[e] += xv * gw[e * H_DIM + h];
  }
#pragma unroll
  for (int e = 0; e < E_NUM; e++) {
    acc[e] += __shfl_xor(acc[e], 16, 32);
    acc[e] += __shfl_xor(acc[e], 8, 32);
    acc[e] += __shfl_xor(acc[e], 4, 32);
    acc[e] += __shfl_xor(acc[e], 2, 32);
    acc[e] += __shfl_xor(acc[e], 1, 32);
  }
  if (lane == 0) {
    int e1 = 0;
#pragma unroll
    for (int e = 1; e < E_NUM; e++) if (acc[e] > acc[e1]) e1 = e;
    int e2 = (e1 == 0) ? 1 : 0;
#pragma unroll
    for (int e = 0; e < E_NUM; e++) if (e != e1 && acc[e] > acc[e2]) e2 = e;
    // softmax + top-2 renorm == 2-way softmax of the two top logits
    float w1 = 1.f / (1.f + __expf(acc[e2] - acc[e1]));
    float w2 = 1.f - w1;
    int p1 = atomicAdd(&cnt[e1], 1);
    tokList[e1 * T_TOK + p1] = t; wtList[e1 * T_TOK + p1] = w1;
    int p2 = atomicAdd(&cnt[e2], 1);
    tokList[e2 * T_TOK + p2] = t; wtList[e2 * T_TOK + p2] = w2;
  }
}

// ---------------- kernel 4: grouped SwiGLU expert FFN, all WMMA ----------------
// block = 16 gathered tokens of one expert; 8 waves.
// phase 1: g,u = x @ Wg/Wu  (M=16, N=I=1024, K=H=2048); A panel staged into LDS
//          16 k-steps at a time with global_load_async_to_lds_b128 (fragment layout),
//          so every wave reads A with two ds_load_b128 instead of redundant vmem.
//          h = silu(g)*u -> LDS, stored directly in phase-2 A-fragment layout.
// phase 2: y = h @ Wd       (M=16, N=H=2048, K=I=1024), out += wt * y (atomic f32)
__global__ __launch_bounds__(256) void k_ffn(const unsigned short* __restrict__ xb,
                                             const unsigned short* __restrict__ wg,
                                             const unsigned short* __restrict__ wu,
                                             const unsigned short* __restrict__ wd,
                                             const int* __restrict__ cnt,
                                             const int* __restrict__ tokList,
                                             const float* __restrict__ wtList,
                                             float* __restrict__ out) {
  const int expert = blockIdx.x >> 9;   // / MAXTILES
  const int tile   = blockIdx.x & (MAXTILES - 1);
  const int c = cnt[expert];
  if (tile * 16 >= c) return;

  __shared__ __align__(16) unsigned short h_lds[(I_DIM / 32) * 32 * 16]; // 32 KB
  __shared__ __align__(16) unsigned short a_lds[16 * 512];               // 16 KB A panel
  __shared__ int   toks[16];
  __shared__ float wts[16];

  const int tid = threadIdx.x;
  if (tid < 16) {
    int slot = tile * 16 + tid;
    if (slot < c) { toks[tid] = tokList[expert * T_TOK + slot];
                    wts[tid]  = wtList[expert * T_TOK + slot]; }
    else          { toks[tid] = 0; wts[tid] = 0.f; }   // pad: weight 0 => no effect
  }
  __syncthreads();

  const int wave = tid >> 5, lane = tid & 31;
  const int lrow = lane & 15, lhi = lane >> 4;
  const int KT1 = H_DIM / 32;  // 64
  const int NT1 = I_DIM / 16;  // 64
  const int KG  = 16;          // k-steps per staged A panel
  const size_t wbaseE = (size_t)expert * NT1 * KT1 * 512;

  // per-thread async-staging descriptors (4 x 16B chunks per thread per group)
  int s_lds[4];
  const unsigned short* s_gp[4];
#pragma unroll
  for (int j = 0; j < 4; ++j) {
    int cid = tid * 4 + j;            // 0..1023
    int ktL = cid >> 6;               // 0..15
    int r6  = cid & 63;
    int Ls  = r6 >> 1;                // lane 0..31
    int hf  = r6 & 1;                 // fragment half
    s_lds[j] = ktL * 1024 + Ls * 32 + hf * 16; // byte offset in a_lds
    s_gp[j]  = xb + (size_t)toks[Ls & 15] * H_DIM + ktL * 32 + hf * 16 + ((Ls >> 4) << 3);
  }
  const unsigned a_base = (unsigned)(size_t)(&a_lds[0]); // low 32 bits = LDS address

  // ---- phase 1: two passes of 512 I-columns (8 waves x 4 tiles of 16) ----
  for (int pass = 0; pass < 2; ++pass) {
    const int ntile0 = pass * 32 + wave * 4;
    v8f Cg[4] = {}, Cu[4] = {};
    for (int kg = 0; kg < KT1; kg += KG) {
      // stage A panel for k-steps [kg, kg+16) : 1024 async b128 copies
#pragma unroll
      for (int j = 0; j < 4; ++j) {
        unsigned ldsoff = a_base + (unsigned)s_lds[j];
        unsigned long long gaddr =
            (unsigned long long)(size_t)(s_gp[j] + (size_t)kg * 32);
        asm volatile("global_load_async_to_lds_b128 %0, %1, off"
                     :: "v"(ldsoff), "v"(gaddr) : "memory");
      }
      asm volatile("s_wait_asynccnt 0x0" ::: "memory");
      __syncthreads();

      for (int ktL = 0; ktL < KG; ++ktL) {
        const int kt = kg + ktL;
        Frag16 a;
        const unsigned short* ap = a_lds + ktL * 512 + lane * 16;
        a.q[0] = *(const u32x4*)(ap);
        a.q[1] = *(const u32x4*)(ap + 8);
        // load ALL B fragments of this k-step first -> 8 independent b128 clauses
        Frag16 bg[4], bu[4];
#pragma unroll
        for (int t2 = 0; t2 < 4; ++t2) {
          const size_t bidx = wbaseE + ((size_t)(ntile0 + t2) * KT1 + kt) * 512 + lane * 16;
          bg[t2].q[0] = *(const u32x4*)(wg + bidx);
          bg[t2].q[1] = *(const u32x4*)(wg + bidx + 8);
          bu[t2].q[0] = *(const u32x4*)(wu + bidx);
          bu[t2].q[1] = *(const u32x4*)(wu + bidx + 8);
        }
        // unconditional WGP/SE-scope prefetch of next k-tile (speculative, in-bounds)
        __builtin_prefetch(wg + wbaseE + ((size_t)ntile0 * KT1 + kt + 1) * 512, 0, 3);
        __builtin_prefetch(wu + wbaseE + ((size_t)ntile0 * KT1 + kt + 1) * 512, 0, 3);
#pragma unroll
        for (int t2 = 0; t2 < 4; ++t2) {
          Cg[t2] = __builtin_amdgcn_wmma_f32_16x16x32_bf16(false, a.v, false, bg[t2].v,
                                                           (short)0, Cg[t2], false, false);
          Cu[t2] = __builtin_amdgcn_wmma_f32_16x16x32_bf16(false, a.v, false, bu[t2].v,
                                                           (short)0, Cu[t2], false, false);
        }
      }
      __syncthreads(); // a_lds re-staged next group
    }
    // silu(g)*u -> LDS, scattered into phase-2 A-fragment layout
#pragma unroll
    for (int t2 = 0; t2 < 4; ++t2) {
      int ncol = (ntile0 + t2) * 16 + lrow;    // I-column this lane owns
      int kt2 = ncol >> 5, kp = ncol & 31;
      int kh = kp & 1, kk = kp >> 1;
      int el2 = (((kk >> 3) * 4 + (kk & 3)) << 1) + kh;
      int g2  = (kk >> 2) & 1;
#pragma unroll
      for (int r = 0; r < 8; ++r) {
        float g = Cg[t2][r], u = Cu[t2][r];
        float s = g * __builtin_amdgcn_rcpf(1.f + __expf(-g)); // fast silu
        int M = r + lhi * 8;
        h_lds[kt2 * 512 + (M + 16 * g2) * 16 + el2] = f32_bf16(s * u);
      }
    }
  }
  __syncthreads();

  // ---- phase 2: each wave owns 16 H-tiles (256 cols), split in two halves ----
  const int NT2 = H_DIM / 16; // 128
  const int KT2 = I_DIM / 32; // 32
  const size_t dbaseE = (size_t)expert * NT2 * KT2 * 512;
  for (int half = 0; half < 2; ++half) {
    const int nt0 = wave * 16 + half * 8;
    v8f C[8] = {};
    for (int kt = 0; kt < KT2; ++kt) {
      Frag16 a;
      const unsigned short* hp = h_lds + kt * 512 + lane * 16;
      a.q[0] = *(const u32x4*)(hp);
      a.q[1] = *(const u32x4*)(hp + 8);
#pragma unroll
      for (int grp = 0; grp < 2; ++grp) {
        Frag16 b[4];
#pragma unroll
        for (int t2 = 0; t2 < 4; ++t2) {
          const size_t bidx =
              dbaseE + ((size_t)(nt0 + grp * 4 + t2) * KT2 + kt) * 512 + lane * 16;
          b[t2].q[0] = *(const u32x4*)(wd + bidx);
          b[t2].q[1] = *(const u32x4*)(wd + bidx + 8);
        }
#pragma unroll
        for (int t2 = 0; t2 < 4; ++t2) {
          C[grp * 4 + t2] = __builtin_amdgcn_wmma_f32_16x16x32_bf16(
              false, a.v, false, b[t2].v, (short)0, C[grp * 4 + t2], false, false);
        }
      }
    }
#pragma unroll
    for (int t2 = 0; t2 < 8; ++t2) {
      int ncol = (nt0 + t2) * 16 + lrow;
#pragma unroll
      for (int r = 0; r < 8; ++r) {
        int M = r + lhi * 8;
        atomicAdd(&out[(size_t)toks[M] * H_DIM + ncol], C[t2][r] * wts[M]);
      }
    }
  }
}

// ---------------- launch ----------------
extern "C" void kernel_launch(void* const* d_in, const int* in_sizes, int n_in,
                              void* d_out, int out_size, void* d_ws, size_t ws_size,
                              hipStream_t stream) {
  (void)in_sizes; (void)n_in; (void)ws_size;
  const float* x   = (const float*)d_in[0]; // [T,H]
  const float* gw  = (const float*)d_in[1]; // [E,H]
  const float* wgf = (const float*)d_in[2]; // [E,H,I]
  const float* wuf = (const float*)d_in[3]; // [E,H,I]
  const float* wdf = (const float*)d_in[4]; // [E,I,H]
  float* out = (float*)d_out;

  // workspace layout (~128.6 MB total; all 256B-aligned)
  char* ws = (char*)d_ws;
  size_t o = 0;
  unsigned short* xb  = (unsigned short*)(ws + o); o += (size_t)T_TOK * H_DIM * 2;
  unsigned short* wgb = (unsigned short*)(ws + o); o += (size_t)E_NUM * H_DIM * I_DIM * 2;
  unsigned short* wub = (unsigned short*)(ws + o); o += (size_t)E_NUM * H_DIM * I_DIM * 2;
  unsigned short* wdb = (unsigned short*)(ws + o); o += (size_t)E_NUM * I_DIM * H_DIM * 2;
  int*   cnt     = (int*)(ws + o);   o += 256;
  int*   tokList = (int*)(ws + o);   o += (size_t)E_NUM * T_TOK * 4;
  float* wtList  = (float*)(ws + o); o += (size_t)E_NUM * T_TOK * 4;

  hipMemsetAsync(d_out, 0, (size_t)out_size * sizeof(float), stream);
  hipMemsetAsync(cnt, 0, 256, stream);

  int nx = T_TOK * H_DIM;
  k_cvt_x<<<nx / 256, 256, 0, stream>>>(x, xb, nx);
  int nw = E_NUM * H_DIM * I_DIM;
  k_swizzle_w<<<nw / 256, 256, 0, stream>>>(wgf, wgb, H_DIM, I_DIM);
  k_swizzle_w<<<nw / 256, 256, 0, stream>>>(wuf, wub, H_DIM, I_DIM);
  k_swizzle_w<<<nw / 256, 256, 0, stream>>>(wdf, wdb, I_DIM, H_DIM);
  k_router<<<T_TOK / 8, 256, 0, stream>>>(x, gw, cnt, tokList, wtList);
  k_ffn<<<E_NUM * MAXTILES, 256, 0, stream>>>(xb, wgb, wub, wdb, cnt, tokList, wtList, out);
}
